// ContrativeNet_58866821759632
// MI455X (gfx1250) — matrix operand order repaired and checked by reading, not compiled, not run on hardware
//
#include <hip/hip_runtime.h>
#include <hip/hip_bf16.h>
#include <math.h>

// ---- problem constants (from reference) ----
#define Nn     51200   // B*NPG
#define Bb     512
#define NPG    100
#define IN_C   100
#define HID    256
#define INDIM  128
#define DEGF   10      // fixed graph degree
#define KK     4       // knn degree
#define PAD100 104     // padded f16 row stride for F=100 (16B-aligned rows)

typedef _Float16 h16;
typedef __attribute__((ext_vector_type(4)))  _Float16 v4h;
typedef __attribute__((ext_vector_type(8)))  _Float16 v8h;
typedef __attribute__((ext_vector_type(16))) _Float16 v16h;
typedef __attribute__((ext_vector_type(8)))  float    v8f;

#define CAT16(lo, hi) __builtin_shufflevector(lo, hi, 0,1,2,3,4,5,6,7,8,9,10,11,12,13,14,15)

// ---- CDNA5 async global->LDS copy (16B per lane), tracked by ASYNCcnt ----
__device__ __forceinline__ void async_ld_b128(const h16* g, h16* l) {
    unsigned loff = (unsigned)(unsigned long long)l;        // LDS byte offset (low 32b of flat addr)
    unsigned long long ga = (unsigned long long)g;
    asm volatile("global_load_async_to_lds_b128 %0, %1, off"
                 :: "v"(loff), "v"(ga) : "memory");
}
__device__ __forceinline__ void wait_async() {
    asm volatile("s_wait_asynccnt 0" ::: "memory");
}

// =====================================================================
// WMMA tiled GEMM: Y[N,Fo](f16) = X[N,Fi](f16) @ W[Fi,Fo](f32)
// 256 threads (8 waves); block tile 128x64; K-step 32.
// Interior K-chunks: A tile staged via global_load_async_to_lds_b128.
// sBt = B tile transposed (B frag = two 16B runs per lane).
// Fis/Fos = padded row strides (16B-aligned); Fi,Fo logical.
// =====================================================================
__global__ __launch_bounds__(256) void k_gemm_wmma(
    const h16* __restrict__ X, const float* __restrict__ W,
    h16* __restrict__ Y, int Fi, int Fis, int Fo, int Fos)
{
    __shared__ __attribute__((aligned(16))) h16 sA [128][40]; // 128 rows x 32 k
    __shared__ __attribute__((aligned(16))) h16 sBt[ 64][40]; // 64 cols x 32 k
    const int tid   = threadIdx.x;
    const int lane  = tid & 31;
    const int wv    = tid >> 5;
    const int khalf = lane >> 4;
    const int m     = lane & 15;
    const int rowBase = blockIdx.x * 128;
    const int colBase = blockIdx.y * 64;

    v8f vzero = {};
    v8f acc[4];
    for (int j = 0; j < 4; ++j) acc[j] = vzero;

    const int ktn = (Fi + 31) >> 5;
    for (int kt = 0; kt < ktn; ++kt) {
        const int k0 = kt * 32;
        if (k0 + 32 <= Fi) {
            // interior chunk: pure f16 copy, async DMA to LDS (2 per thread)
            for (int idx = tid; idx < 128 * 4; idx += 256) {
                int r = idx >> 2, seg = (idx & 3) * 8;
                async_ld_b128(X + (size_t)(rowBase + r) * Fis + k0 + seg,
                              &sA[r][seg]);
            }
        } else {
            // tail chunk: bounded manual staging (Fi%4==0 => group all/none)
            for (int idx = tid; idx < 128 * 8; idx += 256) {
                int r = idx >> 3, g = (idx & 7) * 4;
                int kk = k0 + g;
                v4h p = {};
                if (kk < Fi) p = *(const v4h*)(X + (size_t)(rowBase + r) * Fis + kk);
                *(v4h*)&sA[r][g] = p;
            }
        }
        // stage B transposed: 4 strided f32 loads -> one v4h LDS store
        for (int idx = tid; idx < 64 * 8; idx += 256) {
            int c = idx >> 3, g = (idx & 7) * 4;
            int cc = colBase + c;
            v4h p = {};
            if (cc < Fo) {
                #pragma unroll
                for (int i = 0; i < 4; ++i) {
                    int kk = k0 + g + i;
                    p[i] = (kk < Fi) ? (h16)W[(size_t)kk * Fo + cc] : (h16)0.f;
                }
            }
            *(v4h*)&sBt[c][g] = p;
        }
        wait_async();
        __syncthreads();

        // A fragment: K = khalf*8 + {0..7}, khalf*8 + {16..23}
        const v8h* pa = (const v8h*)&sA[wv * 16 + m][khalf * 8];
        v16h a = CAT16(pa[0], pa[2]);
        #pragma unroll
        for (int j = 0; j < 4; ++j) {
            // B fragment: K = khalf*8 + {0..7}, khalf*8 + {16..23}
            const v8h* pb = (const v8h*)&sBt[j * 16 + m][khalf * 8];
            v16h b = CAT16(pb[0], pb[2]);
            acc[j] = __builtin_amdgcn_wmma_f32_16x16x32_f16(
                false, a, false, b, (short)0, acc[j], false, false);
        }
        __syncthreads();
    }

    // C/D layout: VGPR v: lanes0-15 M=v, lanes16-31 M=v+8; N=lane&15
    for (int j = 0; j < 4; ++j) {
        int col = colBase + j * 16 + m;
        if (col >= Fo) continue;
        #pragma unroll
        for (int v = 0; v < 8; ++v) {
            int row = rowBase + wv * 16 + v + khalf * 8;
            Y[(size_t)row * Fos + col] = (h16)acc[j][v];
        }
    }
}

// =====================================================================
// Per-batch Gram via WMMA: G = Xb @ Xb^T (Xb f16, stride Ds, d logical).
// B[k][n] = Xb[n][k] => BOTH fragments are contiguous runs of row-major sX.
// mode 0: dilated-knn -> 4 src indices/node.  mode 1: affinity output (f32).
// =====================================================================
__global__ __launch_bounds__(256) void k_gram(
    const h16* __restrict__ X, int d, int Ds, int mode,
    int* __restrict__ knn_src, float* __restrict__ aff)
{
    __shared__ __attribute__((aligned(16))) h16 sX[112][40];
    __shared__ float sG[112][113];
    const int b    = blockIdx.x;
    const int tid  = threadIdx.x;
    const int lane = tid & 31, wv = tid >> 5;
    const int khalf = lane >> 4, m = lane & 15;

    // pre-zero pad rows 100..111 once (async path never writes them)
    for (int idx = tid; idx < 12 * 8; idx += 256) {
        int r = NPG + (idx >> 3), g = (idx & 7) * 4;
        v4h z = {};
        *(v4h*)&sX[r][g] = z;
    }

    v8f vzero = {};
    v8f acc[7];
    for (int t = 0; t < 7; ++t) acc[t] = vzero;

    const int ktn = (d + 31) >> 5;
    for (int kt = 0; kt < ktn; ++kt) {
        const int k0 = kt * 32;
        if (k0 + 32 <= d) {
            for (int idx = tid; idx < NPG * 4; idx += 256) {
                int r = idx >> 2, seg = (idx & 3) * 8;
                async_ld_b128(X + ((size_t)b * NPG + r) * Ds + k0 + seg,
                              &sX[r][seg]);
            }
        } else {
            for (int idx = tid; idx < NPG * 8; idx += 256) {
                int r = idx >> 3, g = (idx & 7) * 4;
                int kk = k0 + g;
                v4h p = {};
                if (kk < d) p = *(const v4h*)(X + ((size_t)b * NPG + r) * Ds + kk);
                *(v4h*)&sX[r][g] = p;
            }
        }
        wait_async();
        __syncthreads();
        int tc = 0;
        for (int t = wv; t < 49; t += 8, ++tc) {
            int ti = t / 7, tj = t % 7;
            const v8h* pa = (const v8h*)&sX[ti * 16 + m][khalf * 8];
            v16h a = CAT16(pa[0], pa[2]);
            const v8h* pb = (const v8h*)&sX[tj * 16 + m][khalf * 8];
            v16h bb = CAT16(pb[0], pb[2]);
            acc[tc] = __builtin_amdgcn_wmma_f32_16x16x32_f16(
                false, a, false, bb, (short)0, acc[tc], false, false);
        }
        __syncthreads();
    }
    {
        int tc = 0;
        for (int t = wv; t < 49; t += 8, ++tc) {
            int ti = t / 7, tj = t % 7;
            #pragma unroll
            for (int v = 0; v < 8; ++v)
                sG[ti * 16 + v + khalf * 8][tj * 16 + m] = acc[tc][v];
        }
    }
    __syncthreads();

    if (mode == 0) {
        if (tid < NPG) {
            int i = tid;
            unsigned long long u0 = 0, u1 = 0;
            float gii = sG[i][i];
            int pick[8];
            for (int s = 0; s < 8; ++s) {
                float best = 3.4e38f; int bj = 0;
                for (int j = 0; j < NPG; ++j) {
                    if (j == i) continue;
                    bool used = (j < 64) ? ((u0 >> j) & 1ull)
                                         : ((u1 >> (j - 64)) & 1ull);
                    if (used) continue;
                    float dd = gii + sG[j][j] - 2.f * sG[i][j];
                    if (dd < best) { best = dd; bj = j; }
                }
                if (bj < 64) u0 |= 1ull << bj; else u1 |= 1ull << (bj - 64);
                pick[s] = bj;
            }
            int base = (b * NPG + i) * KK;
            for (int s = 0; s < 4; ++s)
                knn_src[base + s] = b * NPG + pick[2 * s];
        }
    } else {
        for (int idx = tid; idx < NPG * NPG; idx += 256) {
            int i = idx / NPG, j = idx % NPG;
            aff[(size_t)(b * NPG + i) * NPG + j] = (i == j) ? 0.f : sG[i][j];
        }
    }
}

// =====================================================================
// GCN aggregation on f16 activations, f32 accumulate, 4-wide:
// out[n,f] = tanh((xw[n,f] + sum_e xw[src[n*deg+e],f]) / c + bias[f])
// =====================================================================
__global__ void k_gcn_agg4h(const h16* __restrict__ xw,
                            const int* __restrict__ src, int deg,
                            const float* __restrict__ bias, float inv_c,
                            h16* __restrict__ out, int F, int Fs)
{
    const int Fq = F >> 2;
    size_t total = (size_t)Nn * Fq;
    size_t idx = (size_t)blockIdx.x * 256 + threadIdx.x;
    if (idx >= total) return;
    int n = (int)(idx / Fq), f = (int)(idx % Fq) * 4;
    v4h a0 = *(const v4h*)(xw + (size_t)n * Fs + f);
    float s0 = (float)a0[0], s1 = (float)a0[1], s2 = (float)a0[2], s3 = (float)a0[3];
    const int* sp = src + (size_t)n * deg;
    for (int e = 0; e < deg; ++e) {
        v4h v = *(const v4h*)(xw + (size_t)sp[e] * Fs + f);
        s0 += (float)v[0]; s1 += (float)v[1]; s2 += (float)v[2]; s3 += (float)v[3];
    }
    float4 bi = *(const float4*)(bias + f);
    v4h r;
    r[0] = (h16)tanhf(s0 * inv_c + bi.x);
    r[1] = (h16)tanhf(s1 * inv_c + bi.y);
    r[2] = (h16)tanhf(s2 * inv_c + bi.z);
    r[3] = (h16)tanhf(s3 * inv_c + bi.w);
    *(v4h*)(out + (size_t)n * Fs + f) = r;
}

// F == 1 f32 variant (read head)
__global__ void k_gcn_agg1(const float* __restrict__ xw,
                           const int* __restrict__ src, int deg,
                           const float* __restrict__ bias, float inv_c,
                           float* __restrict__ out)
{
    int n = blockIdx.x * 256 + threadIdx.x;
    if (n >= Nn) return;
    float s = xw[n];
    const int* sp = src + (size_t)n * deg;
    for (int e = 0; e < deg; ++e) s += xw[sp[e]];
    out[n] = tanhf(s * inv_c + bias[0]);
}

__global__ void k_add4h(const h16* __restrict__ a, const h16* __restrict__ b,
                        h16* __restrict__ o, int F, int Fs, int leaky)
{
    const int Fq = F >> 2;
    size_t total = (size_t)Nn * Fq;
    size_t idx = (size_t)blockIdx.x * 256 + threadIdx.x;
    if (idx >= total) return;
    int n = (int)(idx / Fq), f = (int)(idx % Fq) * 4;
    v4h x = *(const v4h*)(a + (size_t)n * Fs + f);
    v4h y = *(const v4h*)(b + (size_t)n * Fs + f);
    v4h r;
    #pragma unroll
    for (int i = 0; i < 4; ++i) {
        float v = (float)x[i] + (float)y[i];
        if (leaky) v = v > 0.f ? v : 0.01f * v;
        r[i] = (h16)v;
    }
    *(v4h*)(o + (size_t)n * Fs + f) = r;
}

__global__ void k_read_dot(const h16* __restrict__ x,
                           const float* __restrict__ w, float* __restrict__ o)
{
    int n = blockIdx.x * 256 + threadIdx.x;
    if (n >= Nn) return;
    float s = 0.f;
    const v4h* xr = (const v4h*)(x + (size_t)n * INDIM);
    const float4* wr = (const float4*)w;
    for (int k = 0; k < INDIM / 4; ++k) {
        v4h a = xr[k]; float4 b = wr[k];
        s += (float)a[0] * b.x + (float)a[1] * b.y + (float)a[2] * b.z + (float)a[3] * b.w;
    }
    o[n] = s;
}

__global__ void k_outer(const float* __restrict__ z,
                        const float* __restrict__ w, h16* __restrict__ o)
{
    size_t idx = (size_t)blockIdx.x * 256 + threadIdx.x;
    if (idx >= (size_t)Nn * (INDIM / 4)) return;
    int n = (int)(idx >> 5), f = (int)(idx & 31) * 4;
    float zv = z[n];
    float4 wf = *(const float4*)(w + f);
    v4h r;
    r[0] = (h16)(zv * wf.x); r[1] = (h16)(zv * wf.y);
    r[2] = (h16)(zv * wf.z); r[3] = (h16)(zv * wf.w);
    *(v4h*)(o + (size_t)n * INDIM + f) = r;
}

// convert input x f32[N,100] -> f16[N,104] padded (pads zeroed)
__global__ void k_cvt_pad(const float* __restrict__ in, h16* __restrict__ out)
{
    size_t idx = (size_t)blockIdx.x * 256 + threadIdx.x;
    if (idx >= (size_t)Nn * PAD100) return;
    int n = (int)(idx / PAD100), f = (int)(idx % PAD100);
    out[idx] = (f < IN_C) ? (h16)in[(size_t)n * IN_C + f] : (h16)0.f;
}

// heads: h[B,100] -> sigmoid(h@disW+b), softmax(h@predW+b), mu, log_var,
// z = (eps*exp(0.5lv)+mu)@zdecW + zdecb    (hash-RNG Box-Muller for eps)
__global__ __launch_bounds__(128) void k_head(
    const float* __restrict__ h,
    const float* __restrict__ muW, const float* __restrict__ mub,
    const float* __restrict__ vaW, const float* __restrict__ vab,
    const float* __restrict__ diW, const float* __restrict__ dib,
    const float* __restrict__ prW, const float* __restrict__ prb,
    const float* __restrict__ zdW, const float* __restrict__ zdb,
    float* __restrict__ out_dis, float* __restrict__ out_pred,
    float* __restrict__ out_mu, float* __restrict__ out_lv,
    float* __restrict__ z_out, unsigned seed)
{
    __shared__ float sh[NPG];
    __shared__ float ss[INDIM];
    __shared__ float sl[2];
    int b = blockIdx.x, t = threadIdx.x;
    if (t < NPG) sh[t] = h[b * NPG + t];
    __syncthreads();

    if (t < INDIM) {
        float mu = 0.f, lv = 0.f;
        for (int j = 0; j < NPG; ++j) {
            float hv = sh[j];
            mu += hv * muW[j * INDIM + t];
            lv += hv * vaW[j * INDIM + t];
        }
        mu += mub[t]; lv += vab[t];
        out_mu[b * INDIM + t] = mu;
        out_lv[b * INDIM + t] = lv;
        unsigned i = (unsigned)(b * INDIM + t);
        unsigned x = (seed * 0x9E3779B9u) ^ (i * 0x85EBCA6Bu) ^ 0xC2B2AE35u;
        x ^= x >> 16; x *= 0x7FEB352Du; x ^= x >> 15; x *= 0x846CA68Bu; x ^= x >> 16;
        unsigned y = x * 0x27D4EB2Fu + 0x165667B1u; y ^= y >> 15; y *= 0x2C1B3C6Du; y ^= y >> 12;
        float u1 = ((x >> 8) + 1u) * (1.f / 16777216.f);
        float u2 = (y >> 8) * (1.f / 16777216.f);
        float eps = sqrtf(-2.f * logf(u1)) * cosf(6.28318530718f * u2);
        ss[t] = eps * expf(0.5f * lv) + mu;
    }
    if (t < 2) {
        float dv = 0.f, pv = 0.f;
        for (int j = 0; j < NPG; ++j) {
            float hv = sh[j];
            dv += hv * diW[j * 2 + t];
            pv += hv * prW[j * 2 + t];
        }
        dv += dib[t]; pv += prb[t];
        out_dis[b * 2 + t] = 1.f / (1.f + expf(-dv));
        sl[t] = pv;
    }
    __syncthreads();
    if (t < 2) {
        float mx = fmaxf(sl[0], sl[1]);
        float e0 = expf(sl[0] - mx), e1 = expf(sl[1] - mx);
        out_pred[b * 2 + t] = ((t == 0) ? e0 : e1) / (e0 + e1);
    }
    if (t < NPG) {
        float zv = zdb[t];
        for (int k = 0; k < INDIM; ++k) zv += ss[k] * zdW[k * NPG + t];
        z_out[b * NPG + t] = zv;
    }
}

// =====================================================================
// host-side orchestration
// =====================================================================
static inline int FS(int F) { return (F == 100) ? PAD100 : F; }

static void run_gae(const h16* xh, const int* srcE, const float* const* P,
                    unsigned seed, float* outb,
                    h16* b0, h16* b1, h16* b2, h16* b3,
                    float* hbuf, float* zbuf, float* tbuf, int* knn,
                    hipStream_t s)
{
    const float icF = 1.f / 11.f;   // fixed graph: deg = 10 edges + self loop
    const float icK = 1.f / 5.f;    // knn graph:   deg = 4 edges + self loop
    const size_t AFF = (size_t)Nn * NPG;

    auto gemm = [&](const h16* X, const float* W, h16* Y, int Fi, int Fo) {
        dim3 g(Nn / 128, (unsigned)((Fo + 63) / 64));
        k_gemm_wmma<<<g, 256, 0, s>>>(X, W, Y, Fi, FS(Fi), Fo, FS(Fo));
    };
    auto agg = [&](const h16* xw, const int* sp, int deg, const float* bias,
                   float ic, h16* o, int F) {
        size_t tot = (size_t)Nn * (F >> 2);
        k_gcn_agg4h<<<(unsigned)((tot + 255) / 256), 256, 0, s>>>(
            xw, sp, deg, bias, ic, o, F, FS(F));
    };
    auto add = [&](const h16* a, const h16* c, h16* o, int F, int lk) {
        size_t tot = (size_t)Nn * (F >> 2);
        k_add4h<<<(unsigned)((tot + 255) / 256), 256, 0, s>>>(a, c, o, F, FS(F), lk);
    };
    auto knnk = [&](const h16* X, int d) {
        k_gram<<<Bb, 256, 0, s>>>(X, d, FS(d), 0, knn, (float*)nullptr);
    };

    // ---- encoder stage 0 (enc0: 100 -> 256) ----
    gemm(xh, P[0], b0, IN_C, HID);                // xw shared by x1 and x2
    agg(b0, srcE, DEGF, P[1], icF, b1, HID);      // x1
    knnk(b1, HID);                                // dilated knn on x1
    agg(b0, knn, KK, P[1], icK, b2, HID);         // x2
    add(b1, b2, b3, HID, 1);                      // leaky_relu

    // ---- encoder stage 1 (enc1: 256 -> 128) ----
    gemm(b3, P[2], b0, HID, INDIM);
    agg(b0, srcE, DEGF, P[3], icF, b1, INDIM);
    knnk(b1, INDIM);
    agg(b0, knn, KK, P[3], icK, b2, INDIM);
    add(b1, b2, b3, INDIM, 1);

    // ---- read head: h = gcn(x, read) reshaped (B,100) ----
    k_read_dot<<<(Nn + 255) / 256, 256, 0, s>>>(b3, P[4], tbuf);
    k_gcn_agg1<<<(Nn + 255) / 256, 256, 0, s>>>(tbuf, srcE, DEGF, P[5], icF, hbuf);

    // ---- dis/pred/mu/var + reparameterize + zdec ----
    k_head<<<Bb, 128, 0, s>>>(hbuf,
        P[12], P[13], P[14], P[15], P[16], P[17], P[18], P[19], P[20], P[21],
        outb + AFF, outb + AFF + 1024,
        outb + AFF + 2048, outb + AFF + 2048 + (size_t)Bb * INDIM,
        zbuf, seed);

    // ---- rev: gcn(z[N,1], rev: 1->128) ----
    k_outer<<<(unsigned)(((size_t)Nn * (INDIM / 4) + 255) / 256), 256, 0, s>>>(zbuf, P[6], b0);
    agg(b0, srcE, DEGF, P[7], icF, b3, INDIM);

    // ---- decoder stage 0 (dec0: 128 -> 256) ----
    gemm(b3, P[8], b0, INDIM, HID);
    agg(b0, srcE, DEGF, P[9], icF, b1, HID);
    knnk(b1, HID);
    agg(b0, knn, KK, P[9], icK, b2, HID);
    add(b1, b2, b3, HID, 0);

    // ---- decoder stage 1 (dec1: 256 -> 100) ----
    gemm(b3, P[10], b0, HID, IN_C);
    agg(b0, srcE, DEGF, P[11], icF, b1, IN_C);
    knnk(b1, IN_C);
    agg(b0, knn, KK, P[11], icK, b2, IN_C);
    add(b1, b2, b3, IN_C, 0);

    // ---- affinity: per-batch Xb@Xb^T with zeroed diagonal ----
    k_gram<<<Bb, 256, 0, s>>>(b3, IN_C, FS(IN_C), 1, knn, outb);
}

extern "C" void kernel_launch(void* const* d_in, const int* in_sizes, int n_in,
                              void* d_out, int out_size, void* d_ws, size_t ws_size,
                              hipStream_t stream)
{
    const float* x    = (const float*)d_in[0];
    const int*   edge = (const int*)d_in[1];     // [2,E]: row0=src, row1=dst
    const int*   srcE = edge;                    // dst is repeat(arange(N),10) -> sorted
    // params_hc / params_dis flatten in dict insertion order (W,b per layer)
    const float* Phc[22]; const float* Pds[22];
    for (int i = 0; i < 22; ++i) {
        Phc[i] = (const float*)d_in[6 + i];
        Pds[i] = (const float*)d_in[28 + i];
    }

    char* ws = (char*)d_ws;
    size_t o = 0;
    auto carve = [&](size_t bytes) {
        char* p = ws + o;
        o += (bytes + 255) & ~(size_t)255;
        return p;
    };
    h16*  xh   = (h16*)carve((size_t)Nn * PAD100 * sizeof(h16));
    h16*  b0   = (h16*)carve((size_t)Nn * HID * sizeof(h16));
    h16*  b1   = (h16*)carve((size_t)Nn * HID * sizeof(h16));
    h16*  b2   = (h16*)carve((size_t)Nn * HID * sizeof(h16));
    h16*  b3   = (h16*)carve((size_t)Nn * HID * sizeof(h16));
    float* hbuf = (float*)carve((size_t)Nn * sizeof(float));
    float* zbuf = (float*)carve((size_t)Nn * sizeof(float));
    float* tbuf = (float*)carve((size_t)Nn * sizeof(float));
    int*   knn  = (int*)carve((size_t)Nn * KK * sizeof(int));

    // input f32 -> padded f16 once (shared by both GAE passes)
    k_cvt_pad<<<(unsigned)(((size_t)Nn * PAD100 + 255) / 256), 256, 0, stream>>>(x, xh);

    float* out = (float*)d_out;
    const size_t GAE = (size_t)Nn * NPG + 2 * (size_t)Bb * 2 + 2 * (size_t)Bb * INDIM;

    run_gae(xh, srcE, Phc, 1u, out,       b0, b1, b2, b3, hbuf, zbuf, tbuf, knn, stream);
    run_gae(xh, srcE, Pds, 2u, out + GAE, b0, b1, b2, b3, hbuf, zbuf, tbuf, knn, stream);
}
